// RelayTransformer_38783554683223
// MI455X (gfx1250) — matrix operand order repaired and checked by reading, not compiled, not run on hardware
//
#include <hip/hip_runtime.h>

// ---------------------------------------------------------------------------
// Types for CDNA5 WMMA (wave32, v_wmma_f32_16x16x32_bf16)
// ---------------------------------------------------------------------------
typedef __bf16 bf16;
typedef __bf16 v16bf __attribute__((ext_vector_type(16)));
typedef float  v8f   __attribute__((ext_vector_type(8)));

__device__ __forceinline__ v8f vzero8() {
    v8f z = {0.f, 0.f, 0.f, 0.f, 0.f, 0.f, 0.f, 0.f};
    return z;
}

__device__ __forceinline__ v16bf mkfrag(uint4 lo, uint4 hi) {
    union U { uint4 q[2]; v16bf v; } u;
    u.q[0] = lo; u.q[1] = hi;
    return u.v;
}

__device__ __forceinline__ v8f wmma_bf16(v16bf a, v16bf b, v8f c) {
    // (neg_a, A, neg_b, B, c_mod, C, reuse_a, reuse_b)
    return __builtin_amdgcn_wmma_f32_16x16x32_bf16(false, a, false, b, (short)0, c, false, false);
}

// Model constants
#define DM    1024
#define NH    8
#define DH    128
#define SLEN  2064   // stitched sequence length per batch (16 groups * 129)
#define SVP   2080   // padded key stride for V^T (tail chunk reads 16 extra keys)
#define MROWS 4128   // B * SLEN
#define FFH   4096

// ---------------------------------------------------------------------------
// Weight convert + transpose: W (K x N, f32) -> Wt (N x K, bf16)
// ---------------------------------------------------------------------------
__global__ __launch_bounds__(256) void wt_kernel(const float* __restrict__ W,
                                                 bf16* __restrict__ Wt,
                                                 int K, int N) {
    __shared__ float tile[32][33];
    const int k0 = blockIdx.x * 32, n0 = blockIdx.y * 32;
    const int tx = threadIdx.x & 31, ty = threadIdx.x >> 5;
    for (int r = ty; r < 32; r += 8)
        tile[r][tx] = W[(long)(k0 + r) * N + n0 + tx];
    __syncthreads();
    for (int r = ty; r < 32; r += 8)
        Wt[(long)(n0 + r) * K + k0 + tx] = (bf16)tile[tx][r];
}

// ---------------------------------------------------------------------------
// Build xw (B, 16, 129, D): relay token prepended to each group of 128 tokens
// ---------------------------------------------------------------------------
__global__ __launch_bounds__(256) void build_xw_kernel(const float* __restrict__ x,
                                                       const float* __restrict__ relay,
                                                       float* __restrict__ xw) {
    const long idx = (long)blockIdx.x * 256 + threadIdx.x;
    if (idx >= (long)MROWS * DM) return;
    const int d = (int)(idx & (DM - 1));
    const long r = idx >> 10;
    const int pos = (int)(r % 129);
    const int gb  = (int)(r / 129);   // b*16 + group
    if (pos == 0) {
        xw[idx] = relay[d];
    } else {
        const int b = gb >> 4, grp = gb & 15;
        const long t = (long)grp * 128 + (pos - 1);
        xw[idx] = x[((long)b * 2048 + t) * DM + d];
    }
}

__global__ __launch_bounds__(256) void copy_out_kernel(const float* __restrict__ xw,
                                                       float* __restrict__ out) {
    const long idx = (long)blockIdx.x * 256 + threadIdx.x;
    if (idx >= (long)2 * 2048 * DM) return;
    const int d = (int)(idx & (DM - 1));
    const long r = idx >> 10;          // token row 0..4095
    const int t = (int)(r % 2048);
    const int b = (int)(r / 2048);
    const int grp = t >> 7, i = t & 127;
    out[idx] = xw[(((long)(b * 16 + grp)) * 129 + 1 + i) * DM + d];
}

// ---------------------------------------------------------------------------
// LayerNorm: f32 rows (arbitrary row stride) -> bf16 contiguous rows (D=1024)
// ---------------------------------------------------------------------------
__device__ __forceinline__ float blk_sum256(float v) {
    v += __shfl_xor(v, 16, 32); v += __shfl_xor(v, 8, 32);
    v += __shfl_xor(v,  4, 32); v += __shfl_xor(v, 2, 32);
    v += __shfl_xor(v,  1, 32);
    __shared__ float red[8];
    __shared__ float total;
    const int lane = threadIdx.x & 31, wid = threadIdx.x >> 5;
    if (lane == 0) red[wid] = v;
    __syncthreads();
    float t = (threadIdx.x < 8) ? red[threadIdx.x] : 0.f;
    t += __shfl_xor(t, 4, 32); t += __shfl_xor(t, 2, 32); t += __shfl_xor(t, 1, 32);
    if (threadIdx.x == 0) total = t;
    __syncthreads();
    return total;
}

__global__ __launch_bounds__(256) void ln_kernel(const float* __restrict__ x, long xstride,
                                                 const float* __restrict__ g,
                                                 const float* __restrict__ b,
                                                 bf16* __restrict__ y) {
    const int row = blockIdx.x;
    const float* xr = x + (long)row * xstride;
    const int tid = threadIdx.x;
    float v[4], s = 0.f, ss = 0.f;
#pragma unroll
    for (int i = 0; i < 4; ++i) {
        v[i] = xr[tid + i * 256];
        s += v[i]; ss += v[i] * v[i];
    }
    s  = blk_sum256(s);
    ss = blk_sum256(ss);
    const float mean = s * (1.f / 1024.f);
    const float var  = ss * (1.f / 1024.f) - mean * mean;
    const float rs   = rsqrtf(var + 1e-5f);
    bf16* yr = y + (long)row * DM;
#pragma unroll
    for (int i = 0; i < 4; ++i) {
        const int idx = tid + i * 256;
        yr[idx] = (bf16)((v[i] - mean) * rs * g[idx] + b[idx]);
    }
}

// ---------------------------------------------------------------------------
// Shared WMMA GEMM mainloop: block tile 128x64, 128 threads = 4 waves.
// Wave w owns rows [32w, 32w+32) x all 64 cols -> 8 accumulators,
// 8 WMMAs per k-step with double-buffered LDS (one barrier per k-step).
// Staging kept in named scalar uint4 regs (no arrays/lambdas -> no spills).
// ---------------------------------------------------------------------------
__device__ __forceinline__ void gemm_core(const bf16* __restrict__ A, long lda,
                                          const bf16* __restrict__ Bt, long ldb,
                                          int M, int K, int m0, int n0,
                                          bf16 (*sA)[128][40], bf16 (*sB)[64][40],
                                          v8f acc[2][4]) {
    const int tid = threadIdx.x;
    const int wv = tid >> 5, lane = tid & 31;
    const int lm = lane & 15, lh = lane >> 4;
    const int brow = tid >> 1, bcol = (tid & 1) * 16;
    const uint4 z4 = {0u, 0u, 0u, 0u};
    const int nk = K >> 5;
    const bool arow_ok = (m0 + tid < M);
    const bf16* aptr = A + (long)(m0 + tid) * lda;
    const bf16* bptr = Bt + (long)(n0 + brow) * ldb + bcol;

    uint4 a0 = z4, a1 = z4, a2 = z4, a3 = z4;
    if (arow_ok) {
        a0 = *(const uint4*)(aptr);      a1 = *(const uint4*)(aptr + 8);
        a2 = *(const uint4*)(aptr + 16); a3 = *(const uint4*)(aptr + 24);
    }
    uint4 b0 = *(const uint4*)(bptr), b1 = *(const uint4*)(bptr + 8);
    *(uint4*)&sA[0][tid][0]  = a0; *(uint4*)&sA[0][tid][8]  = a1;
    *(uint4*)&sA[0][tid][16] = a2; *(uint4*)&sA[0][tid][24] = a3;
    *(uint4*)&sB[0][brow][bcol] = b0; *(uint4*)&sB[0][brow][bcol + 8] = b1;
    __syncthreads();

    int cur = 0;
    for (int ks = 0; ks < nk; ++ks) {
        const bool more = (ks + 1 < nk);
        if (more) {
            const int k0 = (ks + 1) << 5;   // prefetch next stage into regs
            if (arow_ok) {
                a0 = *(const uint4*)(aptr + k0);      a1 = *(const uint4*)(aptr + k0 + 8);
                a2 = *(const uint4*)(aptr + k0 + 16); a3 = *(const uint4*)(aptr + k0 + 24);
            }
            b0 = *(const uint4*)(bptr + k0);
            b1 = *(const uint4*)(bptr + k0 + 8);
        }
        const bf16* pa0 = &sA[cur][wv * 32 + lm][0];
        const bf16* pa1 = &sA[cur][wv * 32 + 16 + lm][0];
        const v16bf afr0 = mkfrag(*(const uint4*)(pa0 + lh * 8),
                                  *(const uint4*)(pa0 + 16 + lh * 8));
        const v16bf afr1 = mkfrag(*(const uint4*)(pa1 + lh * 8),
                                  *(const uint4*)(pa1 + 16 + lh * 8));
#pragma unroll
        for (int nt = 0; nt < 4; ++nt) {
            const bf16* pbs = &sB[cur][nt * 16 + lm][lh * 8];
            const v16bf bfg = mkfrag(*(const uint4*)pbs, *(const uint4*)(pbs + 16));
            acc[0][nt] = wmma_bf16(afr0, bfg, acc[0][nt]);
            acc[1][nt] = wmma_bf16(afr1, bfg, acc[1][nt]);
        }
        if (more) {
            const int nxt = cur ^ 1;
            *(uint4*)&sA[nxt][tid][0]  = a0; *(uint4*)&sA[nxt][tid][8]  = a1;
            *(uint4*)&sA[nxt][tid][16] = a2; *(uint4*)&sA[nxt][tid][24] = a3;
            *(uint4*)&sB[nxt][brow][bcol] = b0; *(uint4*)&sB[nxt][brow][bcol + 8] = b1;
            __syncthreads();
            cur = nxt;
        }
    }
}

// Generic GEMM: out = act(A*B + bias) + resid ; f32 and/or bf16 outputs
__global__ __launch_bounds__(128, 2) void gemm_bf16_kernel(
    const bf16* __restrict__ A, long lda, const bf16* __restrict__ Bt,
    const float* __restrict__ bias, const float* __restrict__ resid, long ldr,
    float* __restrict__ outF, long ldoF, bf16* __restrict__ outB, long ldoB,
    int M, int K, int act) {
    __shared__ __attribute__((aligned(16))) bf16 sA[2][128][40];
    __shared__ __attribute__((aligned(16))) bf16 sB[2][64][40];
    v8f acc[2][4] = {{vzero8(), vzero8(), vzero8(), vzero8()},
                     {vzero8(), vzero8(), vzero8(), vzero8()}};
    const int m0 = blockIdx.x * 128, n0 = blockIdx.y * 64;
    gemm_core(A, lda, Bt, (long)K, M, K, m0, n0, sA, sB, acc);

    const int lane = threadIdx.x & 31, wv = threadIdx.x >> 5;
    const int lm = lane & 15, lh = lane >> 4;
#pragma unroll
    for (int sub = 0; sub < 2; ++sub) {
#pragma unroll
        for (int nt = 0; nt < 4; ++nt) {
            const int col = n0 + nt * 16 + lm;
            const float bz = bias ? bias[col] : 0.f;
#pragma unroll
            for (int r = 0; r < 8; ++r) {
                const int row = m0 + wv * 32 + sub * 16 + lh * 8 + r;
                if (row < M) {
                    float c = acc[sub][nt][r] + bz;
                    if (act) c = (c > 0.f) ? c : 0.01f * c;
                    if (resid) c += resid[(long)row * ldr + col];
                    if (outF) outF[(long)row * ldoF + col] = c;
                    if (outB) outB[(long)row * ldoB + col] = (bf16)c;
                }
            }
        }
    }
}

// QKV GEMM: scatter into q/k (b,h,t,dh) bf16 and vT (b,h,dh,t) bf16 (padded keys)
__global__ __launch_bounds__(128, 2) void gemm_qkv_kernel(
    const bf16* __restrict__ A, const bf16* __restrict__ Bt,
    bf16* __restrict__ q, bf16* __restrict__ kk, bf16* __restrict__ vT,
    int M, int K) {
    __shared__ __attribute__((aligned(16))) bf16 sA[2][128][40];
    __shared__ __attribute__((aligned(16))) bf16 sB[2][64][40];
    v8f acc[2][4] = {{vzero8(), vzero8(), vzero8(), vzero8()},
                     {vzero8(), vzero8(), vzero8(), vzero8()}};
    const int m0 = blockIdx.x * 128, n0 = blockIdx.y * 64;
    gemm_core(A, (long)K, Bt, (long)K, M, K, m0, n0, sA, sB, acc);

    const int lane = threadIdx.x & 31, wv = threadIdx.x >> 5;
    const int lm = lane & 15, lh = lane >> 4;
#pragma unroll
    for (int sub = 0; sub < 2; ++sub) {
#pragma unroll
        for (int nt = 0; nt < 4; ++nt) {
            const int col = n0 + nt * 16 + lm;              // 0..3071
            const int which = col >> 10;                    // 0=q 1=k 2=v
            const int c2 = col & 1023, h = c2 >> 7, dh = c2 & 127;
#pragma unroll
            for (int r = 0; r < 8; ++r) {
                const int row = m0 + wv * 32 + sub * 16 + lh * 8 + r;
                if (row < M) {
                    const int b = row / SLEN, t = row % SLEN;
                    const bf16 v = (bf16)acc[sub][nt][r];
                    const long bh = (long)b * NH + h;
                    if (which == 0)      q[(bh * SLEN + t) * DH + dh] = v;
                    else if (which == 1) kk[(bh * SLEN + t) * DH + dh] = v;
                    else                 vT[(bh * DH + dh) * SVP + t] = v;
                }
            }
        }
    }
}

// ---------------------------------------------------------------------------
// Flash-style full attention over stitched sequence (unscaled dots).
// 4 waves / block, each wave owns one (b,h,16-row q-tile); streams 32-key
// chunks (64 full + one 16-key tail). Q*K^T and P*V via WMMA bf16.
// ---------------------------------------------------------------------------
__global__ __launch_bounds__(128, 2) void attn_kernel(const bf16* __restrict__ Q,
                                                      const bf16* __restrict__ Kb,
                                                      const bf16* __restrict__ Vt,
                                                      bf16* __restrict__ O) {
    __shared__ __attribute__((aligned(16))) bf16 sp[4][16][40];
    const int wv = threadIdx.x >> 5, lane = threadIdx.x & 31;
    const int lm = lane & 15, lh = lane >> 4;
    const int gw = blockIdx.x * 4 + wv;
    const int qt = gw % 129, bh = gw / 129;

    // Q fragments: 4 chunks of K-dim (dh) 32
    const long qoff = ((long)bh * SLEN + qt * 16) * DH;
    v16bf aq[4];
#pragma unroll
    for (int c = 0; c < 4; ++c) {
        const bf16* p = Q + qoff + (long)lm * DH + c * 32 + lh * 8;
        aq[c] = mkfrag(*(const uint4*)p, *(const uint4*)(p + 16));
    }

    float rowmax[8], rowsum[8], scl[8];
    v8f oac[8];
#pragma unroll
    for (int r = 0; r < 8; ++r) { rowmax[r] = -1e30f; rowsum[r] = 0.f; }
#pragma unroll
    for (int d = 0; d < 8; ++d) oac[d] = vzero8();

    const bf16* kbase = Kb + (long)bh * SLEN * DH;
    const bf16* vbase = Vt + (long)bh * DH * SVP;

    for (int jc = 0; jc < 65; ++jc) {
        const int j0 = jc * 32;
        const bool full = (jc < 64);
        v8f s0 = vzero8(), s1 = vzero8();
#pragma unroll
        for (int c = 0; c < 4; ++c) {
            const bf16* p0 = kbase + (long)(j0 + lm) * DH + c * 32 + lh * 8;
            s0 = wmma_bf16(aq[c], mkfrag(*(const uint4*)p0, *(const uint4*)(p0 + 16)), s0);
        }
        if (full) {
#pragma unroll
            for (int c = 0; c < 4; ++c) {
                const bf16* p1 = kbase + (long)(j0 + 16 + lm) * DH + c * 32 + lh * 8;
                s1 = wmma_bf16(aq[c], mkfrag(*(const uint4*)p1, *(const uint4*)(p1 + 16)), s1);
            }
        }
        // online softmax: row r lives in lanes {lh*16..lh*16+15}, VGPR r
#pragma unroll
        for (int r = 0; r < 8; ++r) {
            float cm = full ? fmaxf(s0[r], s1[r]) : s0[r];
            cm = fmaxf(cm, __shfl_xor(cm, 1, 32));
            cm = fmaxf(cm, __shfl_xor(cm, 2, 32));
            cm = fmaxf(cm, __shfl_xor(cm, 4, 32));
            cm = fmaxf(cm, __shfl_xor(cm, 8, 32));
            const float mnew = fmaxf(rowmax[r], cm);
            scl[r] = __expf(rowmax[r] - mnew);
            rowmax[r] = mnew;
            const float p0v = __expf(s0[r] - mnew);
            const float p1v = full ? __expf(s1[r] - mnew) : 0.f;
            sp[wv][lh * 8 + r][lm]      = (bf16)p0v;
            sp[wv][lh * 8 + r][16 + lm] = (bf16)p1v;
            float ps = p0v + p1v;
            ps += __shfl_xor(ps, 1, 32);
            ps += __shfl_xor(ps, 2, 32);
            ps += __shfl_xor(ps, 4, 32);
            ps += __shfl_xor(ps, 8, 32);
            rowsum[r] = rowsum[r] * scl[r] + ps;
        }
        __syncthreads();
        // restage P as A-fragment (16 rows x 32 keys)
        const bf16* pp = &sp[wv][lm][0];
        const v16bf pf = mkfrag(*(const uint4*)(pp + lh * 8),
                                *(const uint4*)(pp + 16 + lh * 8));
#pragma unroll
        for (int d = 0; d < 8; ++d) {
#pragma unroll
            for (int r = 0; r < 8; ++r) oac[d][r] *= scl[r];
            const bf16* pv = vbase + (long)(d * 16 + lm) * SVP + j0 + lh * 8;
            oac[d] = wmma_bf16(pf, mkfrag(*(const uint4*)pv, *(const uint4*)(pv + 16)), oac[d]);
        }
    }

    const int b = bh >> 3, h = bh & 7;
#pragma unroll
    for (int r = 0; r < 8; ++r) {
        const float inv = 1.f / rowsum[r];
        const long row = (long)b * SLEN + qt * 16 + lh * 8 + r;
#pragma unroll
        for (int d = 0; d < 8; ++d)
            O[row * DM + h * DH + d * 16 + lm] = (bf16)(oac[d][r] * inv);
    }
}

// ---------------------------------------------------------------------------
// Relay attention: 16 tokens per batch, per (b,h). Negligible FLOPs -> VALU.
// qkv: 32 x 3072 f32 (q|k|v). out: 32 x 1024 bf16.
// ---------------------------------------------------------------------------
__global__ __launch_bounds__(128) void relay_attn_kernel(const float* __restrict__ qkv,
                                                         bf16* __restrict__ out) {
    __shared__ float sc[16][17];
    const int bh = blockIdx.x, b = bh >> 3, h = bh & 7, tid = threadIdx.x;
    for (int e = tid; e < 256; e += 128) {
        const int i = e >> 4, j = e & 15;
        const float* qi = qkv + (long)(b * 16 + i) * 3072 + h * DH;
        const float* kj = qkv + (long)(b * 16 + j) * 3072 + 1024 + h * DH;
        float s = 0.f;
        for (int d = 0; d < DH; ++d) s += qi[d] * kj[d];
        sc[i][j] = s;
    }
    __syncthreads();
    if (tid < 16) {
        float mx = -1e30f;
        for (int j = 0; j < 16; ++j) mx = fmaxf(mx, sc[tid][j]);
        float sm = 0.f;
        for (int j = 0; j < 16; ++j) { const float p = __expf(sc[tid][j] - mx); sc[tid][j] = p; sm += p; }
        const float inv = 1.f / sm;
        for (int j = 0; j < 16; ++j) sc[tid][j] *= inv;
    }
    __syncthreads();
    const int d = tid;  // 0..127
    const float* vb = qkv + (long)(b * 16) * 3072 + 2048 + h * DH + d;
    for (int i = 0; i < 16; ++i) {
        float a = 0.f;
        for (int j = 0; j < 16; ++j) a += sc[i][j] * vb[(long)j * 3072];
        out[(long)(b * 16 + i) * DM + h * DH + d] = (bf16)a;
    }
}

// ---------------------------------------------------------------------------
// Host launch
// ---------------------------------------------------------------------------
extern "C" void kernel_launch(void* const* d_in, const int* in_sizes, int n_in,
                              void* d_out, int out_size, void* d_ws, size_t ws_size,
                              hipStream_t stream) {
    (void)in_sizes; (void)n_in; (void)out_size; (void)ws_size;
    const float* x      = (const float*)d_in[0];
    const float* relay  = (const float*)d_in[1];
    const float* r_ln_g = (const float*)d_in[2];
    const float* r_ln_b = (const float*)d_in[3];
    const float* r_qkv  = (const float*)d_in[4];
    const float* r_ow   = (const float*)d_in[5];
    const float* r_ob   = (const float*)d_in[6];
    const float* l_ln_g = (const float*)d_in[7];
    const float* l_ln_b = (const float*)d_in[8];
    const float* l_qkv  = (const float*)d_in[9];
    const float* l_ow   = (const float*)d_in[10];
    const float* l_ob   = (const float*)d_in[11];
    const float* f_ln_g = (const float*)d_in[12];
    const float* f_ln_b = (const float*)d_in[13];
    const float* f_w1   = (const float*)d_in[14];
    const float* f_b1   = (const float*)d_in[15];
    const float* f_w2   = (const float*)d_in[16];
    const float* f_b2   = (const float*)d_in[17];
    float* out = (float*)d_out;

    char* ws = (char*)d_ws;
    size_t off = 0;
    auto alloc = [&](size_t bytes) -> void* {
        off = (off + 255) & ~(size_t)255;
        void* p = ws + off;
        off += bytes;
        return p;
    };
    float* xw    = (float*)alloc((size_t)MROWS * DM * 4);
    bf16* rqkvT  = (bf16*)alloc((size_t)2 * 3072 * 1024 * 2);
    bf16* rowT   = (bf16*)alloc((size_t)2 * 1024 * 1024 * 2);
    bf16* lqkvT  = (bf16*)alloc((size_t)2 * 3072 * 1024 * 2);
    bf16* lowT   = (bf16*)alloc((size_t)2 * 1024 * 1024 * 2);
    bf16* fw1T   = (bf16*)alloc((size_t)2 * 4096 * 1024 * 2);
    bf16* fw2T   = (bf16*)alloc((size_t)2 * 1024 * 4096 * 2);
    bf16* lnb    = (bf16*)alloc((size_t)MROWS * DM * 2);
    bf16* qb     = (bf16*)alloc((size_t)16 * SLEN * DH * 2);
    bf16* kb     = (bf16*)alloc((size_t)16 * SLEN * DH * 2);
    bf16* vtb    = (bf16*)alloc((size_t)16 * DH * SVP * 2);
    bf16* aob    = (bf16*)alloc((size_t)MROWS * DM * 2);
    bf16* ffh    = (bf16*)alloc((size_t)MROWS * FFH * 2);
    float* rqkvf = (float*)alloc((size_t)32 * 3072 * 4);
    bf16* rlnb   = (bf16*)alloc((size_t)32 * DM * 2);
    bf16* raob   = (bf16*)alloc((size_t)32 * DM * 2);

    // Weight transpose/convert (f32 KxN -> bf16 NxK)
    for (int l = 0; l < 2; ++l) {
        wt_kernel<<<dim3(32, 96), 256, 0, stream>>>(r_qkv + (size_t)l * 1024 * 3072,
                                                    rqkvT + (size_t)l * 3072 * 1024, 1024, 3072);
        wt_kernel<<<dim3(32, 32), 256, 0, stream>>>(r_ow + (size_t)l * 1024 * 1024,
                                                    rowT + (size_t)l * 1024 * 1024, 1024, 1024);
        wt_kernel<<<dim3(32, 96), 256, 0, stream>>>(l_qkv + (size_t)l * 1024 * 3072,
                                                    lqkvT + (size_t)l * 3072 * 1024, 1024, 3072);
        wt_kernel<<<dim3(32, 32), 256, 0, stream>>>(l_ow + (size_t)l * 1024 * 1024,
                                                    lowT + (size_t)l * 1024 * 1024, 1024, 1024);
        wt_kernel<<<dim3(32, 128), 256, 0, stream>>>(f_w1 + (size_t)l * 1024 * 4096,
                                                     fw1T + (size_t)l * 4096 * 1024, 1024, 4096);
        wt_kernel<<<dim3(128, 32), 256, 0, stream>>>(f_w2 + (size_t)l * 4096 * 1024,
                                                     fw2T + (size_t)l * 1024 * 4096, 4096, 1024);
    }
    build_xw_kernel<<<(MROWS * DM + 255) / 256, 256, 0, stream>>>(x, relay, xw);

    const long LDR = 129L * DM;  // relay-token row stride inside xw
    for (int l = 0; l < 2; ++l) {
        // ---- relay attention block (B*n = 32 relay tokens) ----
        ln_kernel<<<32, 256, 0, stream>>>(xw, LDR, r_ln_g + l * DM, r_ln_b + l * DM, rlnb);
        gemm_bf16_kernel<<<dim3(1, 48), 128, 0, stream>>>(
            rlnb, (long)DM, rqkvT + (size_t)l * 3072 * 1024,
            nullptr, nullptr, 0L, rqkvf, 3072L, nullptr, 0L, 32, 1024, 0);
        relay_attn_kernel<<<16, 128, 0, stream>>>(rqkvf, raob);
        gemm_bf16_kernel<<<dim3(1, 16), 128, 0, stream>>>(
            raob, (long)DM, rowT + (size_t)l * 1024 * 1024,
            r_ob + l * DM, xw, LDR, xw, LDR, nullptr, 0L, 32, 1024, 0);

        // ---- stitched full attention (S = 2064 per batch) ----
        ln_kernel<<<MROWS, 256, 0, stream>>>(xw, (long)DM, l_ln_g + l * DM, l_ln_b + l * DM, lnb);
        gemm_qkv_kernel<<<dim3(33, 48), 128, 0, stream>>>(
            lnb, lqkvT + (size_t)l * 3072 * 1024, qb, kb, vtb, MROWS, 1024);
        attn_kernel<<<516, 128, 0, stream>>>(qb, kb, vtb, aob);
        gemm_bf16_kernel<<<dim3(33, 16), 128, 0, stream>>>(
            aob, (long)DM, lowT + (size_t)l * 1024 * 1024,
            l_ob + l * DM, xw, (long)DM, xw, (long)DM, nullptr, 0L, MROWS, 1024, 0);

        // ---- FFN ----
        ln_kernel<<<MROWS, 256, 0, stream>>>(xw, (long)DM, f_ln_g + l * DM, f_ln_b + l * DM, lnb);
        gemm_bf16_kernel<<<dim3(33, 64), 128, 0, stream>>>(
            lnb, (long)DM, fw1T + (size_t)l * 4096 * 1024,
            f_b1 + l * FFH, nullptr, 0L, nullptr, 0L, ffh, (long)FFH, MROWS, 1024, 1);
        gemm_bf16_kernel<<<dim3(33, 16), 128, 0, stream>>>(
            ffh, (long)FFH, fw2T + (size_t)l * 1024 * 4096,
            f_b2 + l * DM, xw, (long)DM, xw, (long)DM, nullptr, 0L, MROWS, 4096, 0);
    }
    copy_out_kernel<<<(2 * 2048 * DM + 255) / 256, 256, 0, stream>>>(xw, out);
}
